// MambaLM_60327110640124
// MI455X (gfx1250) — compile-verified, matrix operand order checked by reading, not run on hardware
//
#include <hip/hip_runtime.h>
#include <hip/hip_bf16.h>
#include <cstdint>
#include <cstddef>

// ---------------- types ----------------
typedef __attribute__((ext_vector_type(16))) __bf16 v16bf;
typedef __attribute__((ext_vector_type(8)))  __bf16 v8bf;
typedef __attribute__((ext_vector_type(8)))  float  v8f;

// ---------------- problem constants ----------------
#define DM   768
#define SM   16
#define TT   1024
#define BB   2
#define MM   (BB*TT)      // 2048 token rows
#define LL   12
#define KCONV 4
#define VV   50257

// ---------------- GEMM tiling ----------------
#define BM 128
#define BN 128
#define BK 32
#define NTHREADS 256      // 8 wave32: 4 (M) x 2 (N) waves, each 32x64 patch

// ====================================================================
// bf16 WMMA GEMM: C[M,N] = A[M,K] * B[K,N] (+ Cadd), f32 in/out.
// f32 operands are staged to LDS with on-the-fly bf16 conversion and
// consumed by v_wmma_f32_16x16x32_bf16 (f32 accumulate).
// Register double-buffering: next K-tile global loads are issued before
// the WMMA block consumes the current LDS tile.
// TRANSB: B given as [N,K] row-major (logits: B = emb).
// ====================================================================
template<bool TRANSB>
__global__ __launch_bounds__(NTHREADS)
void gemm_bf16_wmma(const float* __restrict__ A, const float* __restrict__ Bmat,
                    const float* __restrict__ Cadd, float* __restrict__ C,
                    int M, int N, int Kd)
{
    __shared__ __align__(16) __bf16 As[BM][BK];   // [m][k]
    __shared__ __align__(16) __bf16 Bs[BN][BK];   // [n][k] (transposed stage)

    const int tid  = threadIdx.x;
    const int lane = tid & 31;
    const int wave = tid >> 5;
    const int wm   = wave & 3;          // 4 waves along M (32 rows each)
    const int wn   = wave >> 2;         // 2 waves along N (64 cols each)
    const int mbase = blockIdx.x * BM;
    const int nbase = blockIdx.y * BN;
    const bool fullN = (nbase + BN) <= N;

    const int m0    = wm * 32;
    const int n0    = wn * 64;
    const int row16 = lane & 15;        // A row / B col / C col within 16-tile
    const int half  = lane >> 4;        // lane half selects K groups
    const int kb    = half * 8;         // K base within 16-elem half

    v8f acc[2][4];
    {
        const v8f vzero = {0.f,0.f,0.f,0.f,0.f,0.f,0.f,0.f};
        #pragma unroll
        for (int mi = 0; mi < 2; ++mi)
            #pragma unroll
            for (int ni = 0; ni < 4; ++ni) acc[mi][ni] = vzero;
    }

    // staging registers (16 f32 each for A and B)
    float4 aR[4], bR[4];
    float* arf = reinterpret_cast<float*>(aR);
    float* brf = reinterpret_cast<float*>(bR);

    // ---- global -> regs ----
    auto loadA = [&](int k0) {
        const int ar = tid >> 1;          // 0..127
        const int ks = (tid & 1) * 16;    // 0 / 16
        const int gm = mbase + ar;
        if (gm < M) {
            const float4* ap =
                reinterpret_cast<const float4*>(A + (size_t)gm * Kd + k0 + ks);
            aR[0] = ap[0]; aR[1] = ap[1]; aR[2] = ap[2]; aR[3] = ap[3];
            if (k0 + BK < Kd)   // gfx1250 global_prefetch_b8 one tile ahead
                __builtin_prefetch(A + (size_t)gm * Kd + k0 + BK + ks, 0, 1);
        } else {
            const float4 z4 = {0.f,0.f,0.f,0.f};
            aR[0] = z4; aR[1] = z4; aR[2] = z4; aR[3] = z4;
        }
    };
    auto loadB = [&](int k0) {
        if (!TRANSB) {
            const int kk = tid >> 3;          // 0..31
            const int ns = (tid & 7) * 16;    // 0..112
            const float* bp = Bmat + (size_t)(k0 + kk) * N + nbase + ns;
            if (fullN) {
                const float4* b4 = reinterpret_cast<const float4*>(bp);
                bR[0] = b4[0]; bR[1] = b4[1]; bR[2] = b4[2]; bR[3] = b4[3];
            } else {
                #pragma unroll
                for (int q = 0; q < 16; ++q)
                    brf[q] = (nbase + ns + q < N) ? bp[q] : 0.f;
            }
        } else {
            const int nn = tid >> 1;          // 0..127
            const int ks = (tid & 1) * 16;    // 0 / 16
            const int gn = nbase + nn;
            if (gn < N) {
                const float4* b4 = reinterpret_cast<const float4*>(
                    Bmat + (size_t)gn * Kd + k0 + ks);
                bR[0] = b4[0]; bR[1] = b4[1]; bR[2] = b4[2]; bR[3] = b4[3];
            } else {
                const float4 z4 = {0.f,0.f,0.f,0.f};
                bR[0] = z4; bR[1] = z4; bR[2] = z4; bR[3] = z4;
            }
        }
    };
    // ---- regs -> LDS (with f32 -> bf16 conversion) ----
    auto storeA = [&]() {
        const int ar = tid >> 1;
        const int ks = (tid & 1) * 16;
        #pragma unroll
        for (int q = 0; q < 16; ++q) As[ar][ks + q] = (__bf16)arf[q];
    };
    auto storeB = [&]() {
        if (!TRANSB) {
            const int kk = tid >> 3;
            const int ns = (tid & 7) * 16;
            #pragma unroll
            for (int q = 0; q < 16; ++q) Bs[ns + q][kk] = (__bf16)brf[q];
        } else {
            const int nn = tid >> 1;
            const int ks = (tid & 1) * 16;
            #pragma unroll
            for (int q = 0; q < 16; ++q) Bs[nn][ks + q] = (__bf16)brf[q];
        }
    };

    // ---- prologue: stage first K tile ----
    loadA(0); loadB(0);
    storeA(); storeB();
    __syncthreads();

    for (int k0 = 0;;) {
        const int knext = k0 + BK;
        const bool has_next = knext < Kd;
        if (has_next) { loadA(knext); loadB(knext); }  // overlap with WMMA

        // ---- fragments per ISA 16-bit WMMA layouts; 8 WMMA / wave ----
        v16bf afr[2];
        #pragma unroll
        for (int mi = 0; mi < 2; ++mi) {
            const int ar = m0 + mi*16 + row16;
            v8bf lo = *reinterpret_cast<const v8bf*>(&As[ar][kb]);
            v8bf hi = *reinterpret_cast<const v8bf*>(&As[ar][16 + kb]);
            #pragma unroll
            for (int q = 0; q < 8; ++q) { afr[mi][q] = lo[q]; afr[mi][q+8] = hi[q]; }
        }
        #pragma unroll
        for (int ni = 0; ni < 4; ++ni) {
            const int bcn = n0 + ni*16 + row16;
            v8bf lo = *reinterpret_cast<const v8bf*>(&Bs[bcn][kb]);
            v8bf hi = *reinterpret_cast<const v8bf*>(&Bs[bcn][16 + kb]);
            v16bf bfr;
            #pragma unroll
            for (int q = 0; q < 8; ++q) { bfr[q] = lo[q]; bfr[q+8] = hi[q]; }
            #pragma unroll
            for (int mi = 0; mi < 2; ++mi) {
                acc[mi][ni] = __builtin_amdgcn_wmma_f32_16x16x32_bf16(
                    false, afr[mi], false, bfr, (short)0, acc[mi][ni],
                    false, false);
            }
        }

        if (!has_next) break;
        __syncthreads();          // all waves done reading current LDS tile
        storeA(); storeB();       // publish prefetched tile
        __syncthreads();
        k0 = knext;
    }

    // ---- epilogue: C/D layout VGPR r -> M = r + 8*half, N = lane%16 ----
    #pragma unroll
    for (int mi = 0; mi < 2; ++mi) {
        #pragma unroll
        for (int ni = 0; ni < 4; ++ni) {
            const int gcol = nbase + n0 + ni*16 + row16;
            if (gcol >= N) continue;
            #pragma unroll
            for (int r = 0; r < 8; ++r) {
                const int grow = mbase + m0 + mi*16 + r + 8*half;
                if (grow >= M) continue;
                const size_t idx = (size_t)grow * N + gcol;
                float v = acc[mi][ni][r];
                if (Cadd) v += Cadd[idx];
                C[idx] = v;
            }
        }
    }
}

// ====================================================================
// Embedding gather: x[row] = emb[ids[row]]
// ====================================================================
__global__ __launch_bounds__(256)
void embed_kernel(const int* __restrict__ ids, const float* __restrict__ emb,
                  float* __restrict__ x)
{
    const int row = blockIdx.x;
    const int id  = ids[row];
    const float* er = emb + (size_t)id * DM;
    float* xr = x + (size_t)row * DM;
    for (int d = threadIdx.x; d < DM; d += blockDim.x) xr[d] = er[d];
}

// ====================================================================
// LayerNorm per row (D=768, 256 threads, 3 elems/thread)
// ====================================================================
__global__ __launch_bounds__(256)
void ln_kernel(const float* __restrict__ x, const float* __restrict__ g,
               const float* __restrict__ b, float* __restrict__ out)
{
    __shared__ float s1[256], s2[256];
    const int row = blockIdx.x, tid = threadIdx.x;
    const float* xr = x + (size_t)row * DM;
    float v[3], sum = 0.f, sq = 0.f;
    #pragma unroll
    for (int i = 0; i < 3; ++i) {
        v[i] = xr[tid + i*256]; sum += v[i]; sq += v[i]*v[i];
    }
    s1[tid] = sum; s2[tid] = sq; __syncthreads();
    for (int off = 128; off > 0; off >>= 1) {
        if (tid < off) { s1[tid] += s1[tid+off]; s2[tid] += s2[tid+off]; }
        __syncthreads();
    }
    const float mu  = s1[0] * (1.f/DM);
    const float var = s2[0] * (1.f/DM) - mu*mu;
    const float rs  = rsqrtf(var + 1e-5f);
    float* orow = out + (size_t)row * DM;
    #pragma unroll
    for (int i = 0; i < 3; ++i) {
        const int d = tid + i*256;
        orow[d] = (v[i] - mu) * rs * g[d] + b[d];
    }
}

// ====================================================================
// Depthwise causal conv over time, K=4, reading xc half of xz
// ====================================================================
__global__ __launch_bounds__(256)
void dwconv_kernel(const float* __restrict__ xz, const float* __restrict__ w,
                   float* __restrict__ xc)
{
    const int row = blockIdx.x;
    const int t   = row & (TT - 1);
    const int tid = threadIdx.x;
    #pragma unroll
    for (int i = 0; i < 3; ++i) {
        const int d = tid + i*256;
        float acc = 0.f;
        #pragma unroll
        for (int j = 0; j < KCONV; ++j) {
            const int ts = t - (KCONV - 1) + j;
            if (ts >= 0)
                acc += w[j*DM + d] *
                       xz[(size_t)(row - (KCONV - 1) + j) * (2*DM) + d];
        }
        xc[(size_t)row * DM + d] = acc;
    }
}

// ====================================================================
// SSM-flavored reduction + gate:
//  hs[s] = sum_d xc[d]*(-exp(A_log[d,s])) + Bm[s]*sum_d xc[d]
//  y[d]  = (Cm[d]*sum_s hs[s] + Dp[d]*xc[d]) * silu(z[d])
// ====================================================================
__global__ __launch_bounds__(256)
void scan_kernel(const float* __restrict__ xc, const float* __restrict__ bc,
                 const float* __restrict__ xz, const float* __restrict__ A_log,
                 const float* __restrict__ Dp, float* __restrict__ y)
{
    __shared__ float shx[256];
    __shared__ float shs[SM][256];
    __shared__ float sh_hsum;
    const int row = blockIdx.x, tid = threadIdx.x;
    const float* xr = xc + (size_t)row * DM;

    float px = 0.f, ps[SM];
    #pragma unroll
    for (int s = 0; s < SM; ++s) ps[s] = 0.f;
    float xv[3];
    #pragma unroll
    for (int i = 0; i < 3; ++i) {
        const int d = tid + i*256;
        const float xd = xr[d];
        xv[i] = xd; px += xd;
        const float* ar = A_log + (size_t)d * SM;
        #pragma unroll
        for (int s = 0; s < SM; ++s) ps[s] += xd * (-expf(ar[s]));
    }
    shx[tid] = px;
    #pragma unroll
    for (int s = 0; s < SM; ++s) shs[s][tid] = ps[s];
    __syncthreads();
    for (int off = 128; off > 0; off >>= 1) {
        if (tid < off) shx[tid] += shx[tid+off];
        __syncthreads();
    }
    if (tid < SM) {
        float t = 0.f;
        for (int q = 0; q < 256; ++q) t += shs[tid][q];
        shs[tid][0] = t;
    }
    __syncthreads();
    if (tid == 0) {
        const float sumx = shx[0];
        const float* Bm = bc + (size_t)row * (SM + DM);
        float hsum = 0.f;
        #pragma unroll
        for (int s = 0; s < SM; ++s) hsum += shs[s][0] + Bm[s] * sumx;
        sh_hsum = hsum;
    }
    __syncthreads();
    const float hsum = sh_hsum;
    const float* Cm = bc + (size_t)row * (SM + DM) + SM;
    const float* zr = xz + (size_t)row * (2*DM) + DM;
    float* yr = y + (size_t)row * DM;
    #pragma unroll
    for (int i = 0; i < 3; ++i) {
        const int d = tid + i*256;
        const float zv = zr[d];
        const float silu = zv / (1.f + expf(-zv));
        yr[d] = (Cm[d]*hsum + Dp[d]*xv[i]) * silu;
    }
}

// ====================================================================
// Host orchestration
// ====================================================================
extern "C" void kernel_launch(void* const* d_in, const int* in_sizes, int n_in,
                              void* d_out, int out_size, void* d_ws, size_t ws_size,
                              hipStream_t stream)
{
    const int*   ids   = (const int*)  d_in[0];
    const float* emb   = (const float*)d_in[1];
    const float* Win   = (const float*)d_in[2];
    const float* convw = (const float*)d_in[3];
    const float* Wx    = (const float*)d_in[4];
    const float* A_log = (const float*)d_in[5];
    const float* Dp    = (const float*)d_in[6];
    const float* Wout  = (const float*)d_in[7];
    const float* ln_g  = (const float*)d_in[8];
    const float* ln_b  = (const float*)d_in[9];
    const float* lnf_g = (const float*)d_in[10];
    const float* lnf_b = (const float*)d_in[11];
    float* out = (float*)d_out;

    // workspace (f32): x, h, xz, xc, bc, y  (~44 MB total)
    float* ws    = (float*)d_ws;
    float* xbuf  = ws;
    float* hbuf  = xbuf  + (size_t)MM * DM;
    float* xzbuf = hbuf  + (size_t)MM * DM;
    float* xcbuf = xzbuf + (size_t)MM * 2 * DM;
    float* bcbuf = xcbuf + (size_t)MM * DM;
    float* ybuf  = bcbuf + (size_t)MM * (SM + DM);

    embed_kernel<<<MM, 256, 0, stream>>>(ids, emb, xbuf);

    for (int l = 0; l < LL; ++l) {
        ln_kernel<<<MM, 256, 0, stream>>>(xbuf, ln_g + (size_t)l*DM,
                                          ln_b + (size_t)l*DM, hbuf);
        gemm_bf16_wmma<false><<<dim3(MM/BM, (2*DM)/BN), NTHREADS, 0, stream>>>(
            hbuf, Win + (size_t)l*DM*2*DM, nullptr, xzbuf, MM, 2*DM, DM);
        dwconv_kernel<<<MM, 256, 0, stream>>>(
            xzbuf, convw + (size_t)l*KCONV*DM, xcbuf);
        gemm_bf16_wmma<false><<<dim3(MM/BM, (SM+DM+BN-1)/BN), NTHREADS, 0, stream>>>(
            xcbuf, Wx + (size_t)l*DM*(SM+DM), nullptr, bcbuf, MM, SM+DM, DM);
        scan_kernel<<<MM, 256, 0, stream>>>(
            xcbuf, bcbuf, xzbuf, A_log + (size_t)l*DM*SM,
            Dp + (size_t)l*DM, ybuf);
        gemm_bf16_wmma<false><<<dim3(MM/BM, DM/BN), NTHREADS, 0, stream>>>(
            ybuf, Wout + (size_t)l*DM*DM, xbuf /*residual*/, xbuf, MM, DM, DM);
    }

    ln_kernel<<<MM, 256, 0, stream>>>(xbuf, lnf_g, lnf_b, hbuf);
    // logits = xf @ emb^T : TRANSB, HBM-bound on the 412 MB f32 output
    gemm_bf16_wmma<true><<<dim3(MM/BM, (VV+BN-1)/BN), NTHREADS, 0, stream>>>(
        hbuf, emb, nullptr, out, MM, VV, DM);
}
